// Block_3599182594921
// MI455X (gfx1250) — matrix operand order repaired and checked by reading, not compiled, not run on hardware
//
#include <hip/hip_runtime.h>

// ---------------------------------------------------------------------------
// MI455X (gfx1250, wave32) implementation.
// Heavy GEMMs -> v_wmma_f32_16x16x32_bf16 (f32 inputs converted to bf16 in
// LDS, f32 accumulation), double-buffered LDS, global loads overlapped with
// WMMA. Attention K/V tiles use GLOBAL_LOAD_ASYNC_TO_LDS_B128 when the
// builtin is available. Softmax / adapters / gathers -> VALU.
// ---------------------------------------------------------------------------

typedef __attribute__((ext_vector_type(16))) __bf16 bf16x16;
typedef __attribute__((ext_vector_type(8)))  __bf16 bf16x8;
typedef __attribute__((ext_vector_type(8)))  float  f32x8;
typedef __attribute__((ext_vector_type(4)))  int    v4i;

#define GLOBAL_AS __attribute__((address_space(1)))
#define LDS_AS    __attribute__((address_space(3)))

#if __has_builtin(__builtin_amdgcn_global_load_async_to_lds_b128) && \
    __has_builtin(__builtin_amdgcn_s_wait_asynccnt)
#define USE_ASYNC_LDS 1
#else
#define USE_ASYNC_LDS 0
#endif

__device__ __forceinline__ float geluf(float x) {
    return 0.5f * x * (1.0f + erff(x * 0.70710678118654752f));
}
__device__ __forceinline__ float wred_sum(float v) {
    #pragma unroll
    for (int off = 16; off > 0; off >>= 1) v += __shfl_xor(v, off, 32);
    return v;
}
__device__ __forceinline__ float wred_max(float v) {
    #pragma unroll
    for (int off = 16; off > 0; off >>= 1) v = fmaxf(v, __shfl_xor(v, off, 32));
    return v;
}

// ---------------------------------------------------------------------------
// WMMA GEMM:  out[M,N] = act( A[M,K] @ W[N,K]^T + bias[N] ) (+ res[M,N])
// 256 thr = 8 waves; block tile 128x128x32; wave tile 64x32 (4x2 wmma tiles);
// double-buffered bf16 LDS tiles.
// ---------------------------------------------------------------------------
#define LDS_STRIDE 40   // 32 bf16 + 8 pad (80 bytes/row, keeps 16B alignment)

__device__ __forceinline__ bf16x16 load_fragA(const __bf16* s, int row, int hi) {
    // ISA 7.12.2 16-bit A 16x32: lanes 0-15 K0-7 & K16-23; lanes 16-31 +8
    const bf16x8 lo  = *(const bf16x8*)(s + row * LDS_STRIDE + hi * 8);
    const bf16x8 hi8 = *(const bf16x8*)(s + row * LDS_STRIDE + 16 + hi * 8);
    bf16x16 r;
    #pragma unroll
    for (int i = 0; i < 8; ++i) { r[i] = lo[i]; r[i + 8] = hi8[i]; }
    return r;
}
__device__ __forceinline__ bf16x16 load_fragB(const __bf16* s, int row, int hi) {
    // B 32x16: lanes 0-15 K0-7 & K16-23; lanes 16-31 +8 (same as A)
    const bf16x8 lo  = *(const bf16x8*)(s + row * LDS_STRIDE + hi * 8);
    const bf16x8 hi8 = *(const bf16x8*)(s + row * LDS_STRIDE + 16 + hi * 8);
    bf16x16 r;
    #pragma unroll
    for (int i = 0; i < 8; ++i) { r[i] = lo[i]; r[i + 8] = hi8[i]; }
    return r;
}

template <int BIAS, int ACT, int RES>
__global__ __launch_bounds__(256)
void gemm_bf16_kernel(const float* __restrict__ A, const float* __restrict__ W,
                      const float* __restrict__ bias, const float* __restrict__ res,
                      float* __restrict__ out, int M, int N, int Kd) {
    __shared__ __bf16 sA[2][128 * LDS_STRIDE];
    __shared__ __bf16 sB[2][128 * LDS_STRIDE];

    const int tid = threadIdx.x;
    const int wave = tid >> 5, lane = tid & 31;
    const int lm = lane & 15, hi = lane >> 4;
    const int wm = wave & 1, wn = wave >> 1;          // 2 x 4 wave grid
    const int rowBase = blockIdx.y * 128;
    const int colBase = blockIdx.x * 128;

    // per-thread staging map: 4 float4 groups of A and of W per k-step
    int rr[4], cc[4];
    #pragma unroll
    for (int i = 0; i < 4; ++i) {
        int g = tid + i * 256;                         // 1024 float4 groups
        rr[i] = g >> 3;
        cc[i] = (g & 7) * 4;
    }

    float4 avr[4], bvr[4];
    auto issue_loads = [&](int kb) {
        #pragma unroll
        for (int i = 0; i < 4; ++i) {
            avr[i] = *(const float4*)(A + (size_t)(rowBase + rr[i]) * Kd + kb + cc[i]);
            bvr[i] = *(const float4*)(W + (size_t)(colBase + rr[i]) * Kd + kb + cc[i]);
        }
    };
    auto commit = [&](int buf) {
        #pragma unroll
        for (int i = 0; i < 4; ++i) {
            __bf16* da = &sA[buf][rr[i] * LDS_STRIDE + cc[i]];
            da[0] = (__bf16)avr[i].x; da[1] = (__bf16)avr[i].y;
            da[2] = (__bf16)avr[i].z; da[3] = (__bf16)avr[i].w;
            __bf16* db_ = &sB[buf][rr[i] * LDS_STRIDE + cc[i]];
            db_[0] = (__bf16)bvr[i].x; db_[1] = (__bf16)bvr[i].y;
            db_[2] = (__bf16)bvr[i].z; db_[3] = (__bf16)bvr[i].w;
        }
    };

    f32x8 acc[4][2] = {};

    issue_loads(0);
    commit(0);
    int cur = 0;
    const int nk = Kd >> 5;
    for (int t = 0; t < nk; ++t) {
        __syncthreads();
        if (t + 1 < nk) issue_loads((t + 1) << 5);     // overlap with WMMA below

        bf16x16 af[4], bfr[2];
        #pragma unroll
        for (int mf = 0; mf < 4; ++mf)
            af[mf] = load_fragA(&sA[cur][0], wm * 64 + mf * 16 + lm, hi);
        #pragma unroll
        for (int nf = 0; nf < 2; ++nf)
            bfr[nf] = load_fragB(&sB[cur][0], wn * 32 + nf * 16 + lm, hi);

        #pragma unroll
        for (int mf = 0; mf < 4; ++mf)
            #pragma unroll
            for (int nf = 0; nf < 2; ++nf)
                acc[mf][nf] = __builtin_amdgcn_wmma_f32_16x16x32_bf16(
                    false, af[mf], false, bfr[nf], (short)0, acc[mf][nf], false, false);

        if (t + 1 < nk) commit(cur ^ 1);
        cur ^= 1;
    }

    // epilogue: C/D layout -> lane 0-15: N=lane,M=r ; lane 16-31: N=lane-16,M=8+r
    #pragma unroll
    for (int mf = 0; mf < 4; ++mf) {
        #pragma unroll
        for (int nf = 0; nf < 2; ++nf) {
            int n = colBase + wn * 32 + nf * 16 + lm;
            float bb = BIAS ? bias[n] : 0.0f;
            #pragma unroll
            for (int r = 0; r < 8; ++r) {
                int m = rowBase + wm * 64 + mf * 16 + hi * 8 + r;
                float v = acc[mf][nf][r] + bb;
                if (ACT) v = geluf(v);
                if (RES) v += res[(size_t)m * N + n];
                out[(size_t)m * N + n] = v;
            }
        }
    }
}

// ---------------------------------------------------------------------------
// Elementwise / row-wise kernels
// ---------------------------------------------------------------------------
__global__ void concat_kernel(const float* __restrict__ x, const float* __restrict__ prompt,
                              float* __restrict__ X0, int total) {
    int tid = blockIdx.x * blockDim.x + threadIdx.x;
    if (tid >= total) return;
    int row = tid / 384, c = tid - row * 384;
    int b = row / 528, n = row - b * 528;
    X0[tid] = (n < 16) ? prompt[n * 384 + c]
                       : x[((size_t)b * 512 + (n - 16)) * 384 + c];
}

__global__ __launch_bounds__(256)
void ln_kernel(const float* __restrict__ x, const float* __restrict__ g,
               const float* __restrict__ b, float* __restrict__ y, int M) {
    int row = blockIdx.x * 8 + (threadIdx.x >> 5);
    int lane = threadIdx.x & 31;
    if (row >= M) return;
    const float* xr = x + (size_t)row * 384;
    float v[12], s = 0.f;
    #pragma unroll
    for (int i = 0; i < 12; ++i) { v[i] = xr[lane + 32 * i]; s += v[i]; }
    float mean = wred_sum(s) * (1.0f / 384.0f);
    float s2 = 0.f;
    #pragma unroll
    for (int i = 0; i < 12; ++i) { float d = v[i] - mean; s2 += d * d; }
    float inv = rsqrtf(wred_sum(s2) * (1.0f / 384.0f) + 1e-5f);
    float* yr = y + (size_t)row * 384;
    #pragma unroll
    for (int i = 0; i < 12; ++i) {
        int c = lane + 32 * i;
        yr[c] = (v[i] - mean) * inv * g[c] + b[c];
    }
}

// flash-style attention; wave-per-query-row, 64-key tiles
#define KV_STRIDE 68    // 64 f32 + 4 pad -> 272B rows, 16B aligned for b128

__global__ __launch_bounds__(256)
void attn_kernel(const float* __restrict__ qkv, const float* __restrict__ mask,
                 float* __restrict__ out, int seq, int G, float scale) {
    __shared__ float Ks[64 * KV_STRIDE];
    __shared__ float Vs[64 * KV_STRIDE];
    __shared__ float qs[8 * 64];
    __shared__ float ps[8 * 64];

    const int tid = threadIdx.x, w = tid >> 5, lane = tid & 31;
    const int h = blockIdx.y, bz = blockIdx.z;
    const int q = blockIdx.x * 8 + w;           // seq divisible by 8 in all uses
    const size_t bbase = (size_t)bz * seq;
    const int hcol = h * 64;

    qs[w * 64 + lane]      = qkv[(bbase + q) * 1152 + hcol + lane];
    qs[w * 64 + lane + 32] = qkv[(bbase + q) * 1152 + hcol + lane + 32];

    float mrun = -3e38f, lrun = 0.f, o0 = 0.f, o1 = 0.f;
    const int ntiles = (seq + 63) >> 6;

    for (int kt = 0; kt < ntiles; ++kt) {
        const int kbase = kt * 64;
        int kn = seq - kbase; if (kn > 64) kn = 64;
        __syncthreads();
#if USE_ASYNC_LDS
        // direct global->LDS async copies (ASYNCcnt), 16B per lane per issue
        #pragma unroll
        for (int i = 0; i < 4; ++i) {
            int e = tid + i * 256;              // 1024 float4 chunks per tile
            int r = e >> 4, c = (e & 15) * 4;   // 64 rows x 16 chunks
            if (r < kn) {
                const float* gk = qkv + (bbase + kbase + r) * 1152 + 384 + hcol + c;
                const float* gv = qkv + (bbase + kbase + r) * 1152 + 768 + hcol + c;
                __builtin_amdgcn_global_load_async_to_lds_b128(
                    (GLOBAL_AS v4i*)gk,
                    (LDS_AS v4i*)&Ks[r * KV_STRIDE + c], 0, 0);
                __builtin_amdgcn_global_load_async_to_lds_b128(
                    (GLOBAL_AS v4i*)gv,
                    (LDS_AS v4i*)&Vs[r * KV_STRIDE + c], 0, 0);
            }
        }
        __builtin_amdgcn_s_wait_asynccnt(0);
#else
        #pragma unroll
        for (int i = 0; i < 16; ++i) {
            int e = tid + i * 256;
            int r = e >> 6, c = e & 63;
            if (r < kn) {
                Ks[r * KV_STRIDE + c] = qkv[(bbase + kbase + r) * 1152 + 384 + hcol + c];
                Vs[r * KV_STRIDE + c] = qkv[(bbase + kbase + r) * 1152 + 768 + hcol + c];
            }
        }
#endif
        __syncthreads();

        const int j0 = lane, j1 = lane + 32;
        float s0 = -3e38f, s1 = -3e38f;
        if (j0 < kn) {
            float a = 0.f;
            #pragma unroll 8
            for (int d = 0; d < 64; ++d) a += qs[w * 64 + d] * Ks[j0 * KV_STRIDE + d];
            s0 = a * scale;
            if (mask && q < G && (kbase + j0) < G)
                s0 += mask[((size_t)bz * G + q) * G + kbase + j0] * (-100000.0f);
        }
        if (j1 < kn) {
            float a = 0.f;
            #pragma unroll 8
            for (int d = 0; d < 64; ++d) a += qs[w * 64 + d] * Ks[j1 * KV_STRIDE + d];
            s1 = a * scale;
            if (mask && q < G && (kbase + j1) < G)
                s1 += mask[((size_t)bz * G + q) * G + kbase + j1] * (-100000.0f);
        }
        float mnew = fmaxf(mrun, wred_max(fmaxf(s0, s1)));
        float p0 = (j0 < kn) ? __expf(s0 - mnew) : 0.f;
        float p1 = (j1 < kn) ? __expf(s1 - mnew) : 0.f;
        float lt = wred_sum(p0 + p1);
        float corr = __expf(mrun - mnew);
        lrun = lrun * corr + lt;
        o0 *= corr; o1 *= corr;
        mrun = mnew;
        ps[w * 64 + lane] = p0;
        ps[w * 64 + lane + 32] = p1;
        __syncthreads();
        for (int j = 0; j < kn; ++j) {
            float p = ps[w * 64 + j];
            o0 += p * Vs[j * KV_STRIDE + lane];
            o1 += p * Vs[j * KV_STRIDE + lane + 32];
        }
    }
    float inv = 1.0f / lrun;
    out[(bbase + q) * 384 + hcol + lane]      = o0 * inv;
    out[(bbase + q) * 384 + hcol + lane + 32] = o1 * inv;
}

__global__ void gather_kernel(const float* __restrict__ src, const int* __restrict__ idx,
                              float* __restrict__ dst, int nrows) {
    int tid = blockIdx.x * blockDim.x + threadIdx.x;
    int total = nrows * 96;
    if (tid >= total) return;
    int r = tid / 96, qv = tid - r * 96;
    int s = idx[r];
    *(float4*)(dst + (size_t)r * 384 + qv * 4) =
        *(const float4*)(src + (size_t)s * 384 + qv * 4);
}

__global__ void knnreduce_kernel(const float* __restrict__ xn, const float* __restrict__ cent,
                                 const float* __restrict__ bg, const float* __restrict__ bb,
                                 float* __restrict__ vis, int BS) {
    int tid = blockIdx.x * blockDim.x + threadIdx.x;
    int total = BS * 384;
    if (tid >= total) return;
    int bs = tid / 384, c = tid - bs * 384;
    const float* p = xn + (size_t)bs * 32 * 384 + c;
    float mx = -3e38f, sm = 0.f;
    #pragma unroll 8
    for (int k = 0; k < 32; ++k) { float v = p[(size_t)k * 384]; mx = fmaxf(mx, v); sm += v; }
    float lc = mx + sm * (1.0f / 32.0f);
    lc = geluf(lc * bg[c] * rsqrtf(1.0f + 1e-5f) + bb[c]);
    vis[tid] = lc + 0.4f * cent[tid];
}

__global__ __launch_bounds__(256)
void weight_kernel(const float* __restrict__ c1, const float* __restrict__ c2,
                   float* __restrict__ wgt, int BG) {
    int row = blockIdx.x * 8 + (threadIdx.x >> 5);
    int lane = threadIdx.x & 31;
    if (row >= BG) return;
    int b = row >> 9;
    float x1 = c1[row * 3], y1 = c1[row * 3 + 1], z1 = c1[row * 3 + 2];
    float n1 = x1 * x1 + y1 * y1 + z1 * z1;
    float r[4], tot = 0.f;
    #pragma unroll
    for (int i = 0; i < 4; ++i) {
        int s = lane + i * 32;
        const float* p = c2 + ((size_t)b * 128 + s) * 3;
        float x2 = p[0], y2 = p[1], z2 = p[2];
        float d = n1 + x2 * x2 + y2 * y2 + z2 * z2 - 2.0f * (x1 * x2 + y1 * y2 + z1 * z2);
        r[i] = 1.0f / (d + 1e-8f);
        tot += r[i];
    }
    float inv = 1.0f / wred_sum(tot);
    #pragma unroll
    for (int i = 0; i < 4; ++i) wgt[(size_t)row * 128 + lane + i * 32] = r[i] * inv;
}

__global__ void interp_kernel(const float* __restrict__ wgt, const float* __restrict__ vis,
                              const float* __restrict__ X2, float* __restrict__ nx, int BG) {
    int tid = blockIdx.x * blockDim.x + threadIdx.x;
    int total = BG * 384;
    if (tid >= total) return;
    int row = tid / 384, c = tid - row * 384;
    int b = row >> 9, n = row & 511;
    const float* wr = wgt + (size_t)row * 128;
    const float* vb = vis + (size_t)b * 128 * 384 + c;
    float acc = 0.f;
    #pragma unroll 4
    for (int s = 0; s < 128; ++s) acc += wr[s] * vb[(size_t)s * 384];
    float xg = X2[((size_t)b * 528 + 16 + n) * 384 + c];
    nx[tid] = xg + 0.4f * acc;
}

// adapter down-projection (rank 16): wave-per-row
__global__ __launch_bounds__(256)
void adapterA_kernel(const float* __restrict__ x, const float* __restrict__ dw,
                     const float* __restrict__ db, float* __restrict__ tmp, int M) {
    int row = blockIdx.x * 8 + (threadIdx.x >> 5);
    int lane = threadIdx.x & 31;
    if (row >= M) return;
    int r = lane & 15, half = lane >> 4;
    const float* xr = x + (size_t)row * 384 + half * 192;
    const float* wr = dw + (size_t)r * 384 + half * 192;
    float a = 0.f;
    #pragma unroll 8
    for (int k = 0; k < 192; ++k) a += xr[k] * wr[k];
    a += __shfl_xor(a, 16, 32);
    if (half == 0) tmp[(size_t)row * 16 + r] = geluf(a + db[r]);
}

// adapter up-projection + residual combine
template <int GATED>
__global__ void adapterB_kernel(const float* __restrict__ tmp, const float* __restrict__ uw,
                                const float* __restrict__ ub, const float* __restrict__ xin,
                                const float* __restrict__ base, const float* __restrict__ gatep,
                                float* __restrict__ out, int M) {
    int tid = blockIdx.x * blockDim.x + threadIdx.x;
    int total = M * 384;
    if (tid >= total) return;
    int row = tid / 384, c = tid - row * 384;
    const float* tr = tmp + (size_t)row * 16;
    const float* wr = uw + (size_t)c * 16;
    float s = 0.f;
    #pragma unroll
    for (int r = 0; r < 16; ++r) s += tr[r] * wr[r];
    float v = s + ub[c] + xin[tid];
    if (GATED) v = gatep[0] * v + base[tid];
    out[tid] = v;
}

// ---------------------------------------------------------------------------
// Host orchestration
// ---------------------------------------------------------------------------
extern "C" void kernel_launch(void* const* d_in, const int* in_sizes, int n_in,
                              void* d_out, int out_size, void* d_ws, size_t ws_size,
                              hipStream_t stream) {
    (void)in_sizes; (void)n_in; (void)out_size; (void)ws_size;

    const float* x        = (const float*)d_in[0];
    const float* mask     = (const float*)d_in[1];
    const float* center1  = (const float*)d_in[2];
    const float* center2  = (const float*)d_in[3];
    /* neighborhood d_in[4] unused by the reference math */
    const float* prompt   = (const float*)d_in[5];
    const float* n1_g = (const float*)d_in[6],  *n1_b = (const float*)d_in[7];
    const float* qkv_w = (const float*)d_in[8], *proj_w = (const float*)d_in[9], *proj_b = (const float*)d_in[10];
    const float* n2_g = (const float*)d_in[11], *n2_b = (const float*)d_in[12];
    const float* fc1_w = (const float*)d_in[13], *fc1_b = (const float*)d_in[14];
    const float* fc2_w = (const float*)d_in[15], *fc2_b = (const float*)d_in[16];
    const float* ad_gate = (const float*)d_in[17];
    const float* ad_dw = (const float*)d_in[18], *ad_db = (const float*)d_in[19];
    const float* ad_uw = (const float*)d_in[20], *ad_ub = (const float*)d_in[21];
    const float* ad1_dw = (const float*)d_in[22], *ad1_db = (const float*)d_in[23];
    const float* ad1_uw = (const float*)d_in[24], *ad1_ub = (const float*)d_in[25];
    const float* bn_g = (const float*)d_in[26], *bn_b = (const float*)d_in[27];
    const float* a1_qkv_w = (const float*)d_in[28];
    const float* a1_proj_w = (const float*)d_in[29], *a1_proj_b = (const float*)d_in[30];
    const float* n3_g = (const float*)d_in[31], *n3_b = (const float*)d_in[32];
    const int* idx = (const int*)d_in[33];
    const int* center_idx = (const int*)d_in[34];

    const int B = 32, G = 512, S = 128, C = 384, H = 6;
    const int N1 = G + 16;                // 528
    const long M1 = (long)B * N1;         // 16896
    const long M2 = (long)B * S * 32;     // 131072
    const long BS = (long)B * S;          // 4096
    const long BG = (long)B * G;          // 16384
    const int C3 = 3 * C, H4 = 4 * C;

    // workspace bump allocator
    char* wsb = (char*)d_ws;
    size_t off = 0;
    auto alloc = [&](size_t elems) -> float* {
        float* p = (float*)(wsb + off);
        off = (off + elems * sizeof(float) + 255) & ~(size_t)255;
        return p;
    };
    float* X0   = alloc(M1 * C);
    float* XLN  = alloc(M1 * C);          // reused: LN1 input to qkv, then LN2
    float* QKV1 = alloc(M1 * C3);
    float* AO1  = alloc(M1 * C);
    float* X1   = alloc(M1 * C);
    float* FC1  = alloc(M1 * H4);
    float* XFN  = alloc(M1 * C);
    float* TMP  = alloc(M1 * 16);
    float* X2   = alloc(M1 * C);
    float* GAT  = alloc(M2 * C);
    float* CENT = alloc(BS * C);
    float* NLN  = alloc(M2 * C);
    float* QKV2 = alloc(M2 * C3);
    float* AO2  = NLN;                    // reuse (NLN dead after QKV2 GEMM)
    float* XN2  = QKV2;                   // reuse (QKV2 dead after attention)
    float* VIS  = alloc(BS * C);
    float* WGT  = alloc(BG * S);
    float* NEWX = alloc(BG * C);
    float* TMP2 = alloc(BG * 16);

    const float scale = 0.125f;           // 64^-0.5

    // ---- stage 1: prompt-attached transformer block ----
    concat_kernel<<<(unsigned)((M1 * C + 255) / 256), 256, 0, stream>>>(x, prompt, X0, (int)(M1 * C));
    ln_kernel<<<(unsigned)(M1 / 8), 256, 0, stream>>>(X0, n1_g, n1_b, XLN, (int)M1);
    gemm_bf16_kernel<0, 0, 0><<<dim3(C3 / 128, (unsigned)(M1 / 128)), 256, 0, stream>>>(
        XLN, qkv_w, nullptr, nullptr, QKV1, (int)M1, C3, C);
    attn_kernel<<<dim3(N1 / 8, H, B), 256, 0, stream>>>(QKV1, mask, AO1, N1, G, scale);
    gemm_bf16_kernel<1, 0, 1><<<dim3(C / 128, (unsigned)(M1 / 128)), 256, 0, stream>>>(
        AO1, proj_w, proj_b, X0, X1, (int)M1, C, C);
    ln_kernel<<<(unsigned)(M1 / 8), 256, 0, stream>>>(X1, n2_g, n2_b, XLN, (int)M1);
    gemm_bf16_kernel<1, 1, 0><<<dim3(H4 / 128, (unsigned)(M1 / 128)), 256, 0, stream>>>(
        XLN, fc1_w, fc1_b, nullptr, FC1, (int)M1, H4, C);
    gemm_bf16_kernel<1, 0, 0><<<dim3(C / 128, (unsigned)(M1 / 128)), 256, 0, stream>>>(
        FC1, fc2_w, fc2_b, nullptr, XFN, (int)M1, C, H4);
    adapterA_kernel<<<(unsigned)(M1 / 8), 256, 0, stream>>>(XFN, ad_dw, ad_db, TMP, (int)M1);
    adapterB_kernel<1><<<(unsigned)((M1 * C + 255) / 256), 256, 0, stream>>>(
        TMP, ad_uw, ad_ub, XFN, X1, ad_gate, X2, (int)M1);

    // ---- stage 2: gathered local attention over knn groups ----
    gather_kernel<<<(unsigned)((M2 * 96 + 255) / 256), 256, 0, stream>>>(X2, idx, GAT, (int)M2);
    gather_kernel<<<(unsigned)((BS * 96 + 255) / 256), 256, 0, stream>>>(X2, center_idx, CENT, (int)BS);
    ln_kernel<<<(unsigned)(M2 / 8), 256, 0, stream>>>(GAT, n3_g, n3_b, NLN, (int)M2);
    gemm_bf16_kernel<0, 0, 0><<<dim3(C3 / 128, (unsigned)(M2 / 128)), 256, 0, stream>>>(
        NLN, a1_qkv_w, nullptr, nullptr, QKV2, (int)M2, C3, C);
    attn_kernel<<<dim3(32 / 8, H, (unsigned)BS), 256, 0, stream>>>(QKV2, nullptr, AO2, 32, G, scale);
    gemm_bf16_kernel<1, 0, 1><<<dim3(C / 128, (unsigned)(M2 / 128)), 256, 0, stream>>>(
        AO2, a1_proj_w, a1_proj_b, GAT, XN2, (int)M2, C, C);

    // ---- stage 3: pooling, inverse-distance interpolation, final adapter ----
    knnreduce_kernel<<<(unsigned)((BS * C + 255) / 256), 256, 0, stream>>>(
        XN2, CENT, bn_g, bn_b, VIS, (int)BS);
    weight_kernel<<<(unsigned)(BG / 8), 256, 0, stream>>>(center1, center2, WGT, (int)BG);
    interp_kernel<<<(unsigned)((BG * C + 255) / 256), 256, 0, stream>>>(WGT, VIS, X2, NEWX, (int)BG);
    adapterA_kernel<<<(unsigned)(BG / 8), 256, 0, stream>>>(NEWX, ad1_dw, ad1_db, TMP2, (int)BG);
    adapterB_kernel<0><<<(unsigned)((BG * C + 255) / 256), 256, 0, stream>>>(
        TMP2, ad1_uw, ad1_ub, NEWX, nullptr, nullptr, (float*)d_out, (int)BG);
}
